// SpMiddleResNetFHDLargeKernel_23338852287175
// MI455X (gfx1250) — matrix operand order repaired and checked by reading, not compiled, not run on hardware
//
#include <hip/hip_runtime.h>
#include <hip/hip_bf16.h>
#include <stdint.h>

// ---------------------------------------------------------------------------
// SpMiddleResNetFHD for MI455X (gfx1250).
// Dense NDHWC bf16 activations (channel-padded), implicit-GEMM conv via
// v_wmma_f32_16x16x32_bf16 (wave32), f32 accumulate, fused
// bias/BN/mask/residual/ReLU epilogue. Stage-1 tensors (52MB bf16) are
// L2-resident (192MB), so the pipeline is compute-bound on the WMMA units.
// All per-lane memory ops are forced to address-space(1) so they lower to
// global_load/global_store (LOADcnt/STOREcnt only), never flat (DScnt).
// ---------------------------------------------------------------------------

typedef __attribute__((ext_vector_type(16))) __bf16        v16bf;
typedef __attribute__((ext_vector_type(8)))  float         v8f;
typedef __attribute__((ext_vector_type(4)))  unsigned int  u32x4;  // POD 16B

// ---- global-address-space access helpers (avoid flat_* lowering) ----------
__device__ __forceinline__ u32x4 ldg_u4(const unsigned short* p) {
    return *(const __attribute__((address_space(1))) u32x4*)p;
}
__device__ __forceinline__ float ldg_f(const float* p) {
    return *(const __attribute__((address_space(1))) float*)p;
}
__device__ __forceinline__ unsigned short ldg_u16(const unsigned short* p) {
    return *(const __attribute__((address_space(1))) unsigned short*)p;
}
__device__ __forceinline__ void stg_u16(unsigned short* p, unsigned short v) {
    *(__attribute__((address_space(1))) unsigned short*)p = v;
}
__device__ __forceinline__ void stg_f(float* p, float v) {
    *(__attribute__((address_space(1))) float*)p = v;
}

__device__ __forceinline__ unsigned short f2bf(float f) {
    unsigned int u = __float_as_uint(f);
    u += 0x7fffu + ((u >> 16) & 1u);   // round-to-nearest-even
    return (unsigned short)(u >> 16);
}
__device__ __forceinline__ float bf2f(unsigned short h) {
    return __uint_as_float(((unsigned int)h) << 16);
}

union Frag {
    v16bf v;
    u32x4 q[2];
};

struct ConvP {
    const unsigned short* in;     // bf16 NDHWC, channel stride Cs_in
    const unsigned short* wt;     // bf16 [taps][Co][CiPad]
    unsigned short*       out;    // bf16 NDHWC (if out_f32 == nullptr)
    float*                out_f32;// f32 NCDHW (final layer)
    const float*          mask;   // f32 at output resolution
    const unsigned short* resid;  // bf16 NDHWC, same dims as out (or null)
    const float*          scale;  // per-Co: gamma/sqrt(var+eps)
    const float*          shift;  // per-Co: beta + (bias - mean)*scale
    int Din, Hin, Win, Cs_in;
    int Dout, Hout, Wout, Cs_out;
    int kd, kh, kw, sd, sh, sw, pd, ph, pw;
    int Co, CiPad, relu;
};

// One wave computes a 16(x-positions) x 16*COG (out-channels) output tile.
// A layout (16-bit A 16x32): lane = half*16 + m; elements 0..7 = K half*8..+7,
// elements 8..15 = K 16+half*8..+7  -> two contiguous 8-channel (16B) runs.
// B layout (32x16): col N=lane&15; elements 0..7 = K half*8..+7,
//   elements 8..15 = K 16+half*8..+7 -> two 16B runs from [tap][co][ci] weights.
// C/D layout: vgpr r, lane -> (m = r + 8*(lane>>4), n = lane&15).
template <int COG>
__global__ void __launch_bounds__(128) conv3d_wmma(ConvP p) {
    const int lane = threadIdx.x;        // 0..31
    const int wv   = threadIdx.y;        // 0..3 (4 waves / block)
    const int tilesX = (p.Wout + 15) >> 4;
    const int tiles  = tilesX * p.Hout * p.Dout;
    const int tile   = blockIdx.x * 4 + wv;       // wave-uniform
    if (tile >= tiles) return;                    // uniform exit: EXEC stays full
    const int tx = tile % tilesX;
    const int t2 = tile / tilesX;
    const int y  = t2 % p.Hout;
    const int z  = t2 / p.Hout;
    const int x0 = tx << 4;
    const int coBase = blockIdx.y * (16 * COG);

    const int mrow = lane & 15;
    const int half = lane >> 4;

    const int xo     = x0 + mrow;                 // A-row output x position
    const int xiBase = xo * p.sw - p.pw;
    const int zi0    = z * p.sd - p.pd;
    const int yi0    = y * p.sh - p.ph;

    v8f acc[COG];
#pragma unroll
    for (int g = 0; g < COG; ++g)
#pragma unroll
        for (int r = 0; r < 8; ++r) acc[g][r] = 0.f;

    const int   chunks = p.CiPad >> 5;
    const u32x4 z4 = (u32x4)(0u);

    for (int dz = 0; dz < p.kd; ++dz) {
        const int zi = zi0 + dz;
        if ((unsigned)zi >= (unsigned)p.Din) continue;         // uniform skip
        for (int dy = 0; dy < p.kh; ++dy) {
            const int yi = yi0 + dy;
            if ((unsigned)yi >= (unsigned)p.Hin) continue;     // uniform skip
            const unsigned short* rowBase =
                p.in + ((size_t)zi * p.Hin + yi) * (size_t)p.Win * p.Cs_in;
            for (int dx = 0; dx < p.kw; ++dx) {
                const int  xi  = xiBase + dx;
                const bool inX = ((unsigned)xi < (unsigned)p.Win);
                const int  tap = (dz * p.kh + dy) * p.kw + dx;
                // clamped, always-in-bounds base (row exists; xi clamped)
                const unsigned short* aPos =
                    rowBase + (size_t)(inX ? xi : 0) * p.Cs_in;
                const unsigned short* wTap =
                    p.wt + (size_t)tap * p.Co * p.CiPad + (size_t)half * 8;
                __builtin_prefetch(wTap, 0, 1);   // global_prefetch_b8
                for (int ch = 0; ch < chunks; ++ch) {
                    const int  c0 = (ch << 5) + half * 8;
                    const int  c1 = c0 + 16;
                    const bool v0 = inX && (c0 + 8) <= p.Cs_in;
                    const bool v1 = inX && (c1 + 8) <= p.Cs_in;
                    // unconditional loads from clamped addresses, then select
                    // on DATA (v_cndmask) -> keeps global_load_b128 lowering
                    u32x4 t0 = ldg_u4(aPos + ((c0 + 8) <= p.Cs_in ? c0 : 0));
                    u32x4 t1 = ldg_u4(aPos + ((c1 + 8) <= p.Cs_in ? c1 : 0));
                    Frag a;
                    a.q[0] = v0 ? t0 : z4;
                    a.q[1] = v1 ? t1 : z4;
#pragma unroll
                    for (int g = 0; g < COG; ++g) {
                        const int co = coBase + g * 16 + mrow;
                        Frag b;
                        const unsigned short* wp =
                            wTap + (size_t)co * p.CiPad + (ch << 5);
                        b.q[0] = ldg_u4(wp);
                        b.q[1] = ldg_u4(wp + 16);
                        acc[g] = __builtin_amdgcn_wmma_f32_16x16x32_bf16(
                            false, a.v, false, b.v, (short)0, acc[g],
                            false, false);
                    }
                }
            }
        }
    }

    // Fused epilogue: BN(+bias) -> *mask -> +residual -> ReLU -> store.
    const size_t rowOut = ((size_t)z * p.Hout + y) * (size_t)p.Wout;
#pragma unroll
    for (int g = 0; g < COG; ++g) {
        const int   co = coBase + g * 16 + mrow;
        const float sc = ldg_f(p.scale + co);
        const float sh = ldg_f(p.shift + co);
#pragma unroll
        for (int r = 0; r < 8; ++r) {
            const int mm = r + 8 * half;
            const int xw = x0 + mm;
            if (xw >= p.Wout) continue;
            const size_t sp = rowOut + xw;
            float v = acc[g][r] * sc + sh;
            v *= ldg_f(p.mask + sp);
            if (p.resid) v += bf2f(ldg_u16(p.resid + sp * (size_t)p.Cs_out + co));
            if (p.relu && v < 0.f) v = 0.f;
            if (p.out_f32) {
                // NCDHW for the final (c, d, h, w) -> (c*D+d, h, w) reshape
                stg_f(p.out_f32 +
                      (((size_t)co * p.Dout + z) * p.Hout + y) * p.Wout + xw, v);
            } else {
                stg_u16(p.out + sp * (size_t)p.Cs_out + co, f2bf(v));
            }
        }
    }
}

// ---- helper kernels --------------------------------------------------------

__global__ void zero_k(unsigned int* p, size_t n) {
    size_t i = (size_t)blockIdx.x * blockDim.x + threadIdx.x;
    if (i < n) p[i] = 0u;
}

__global__ void scatter_k(const float* vf, const int* coors,
                          unsigned short* dense, float* mask,
                          int H, int W, int Cs, int n) {
    int i = blockIdx.x * blockDim.x + threadIdx.x;
    if (i >= n) return;
    const int z = coors[4 * i + 1];
    const int y = coors[4 * i + 2];
    const int x = coors[4 * i + 3];
    size_t sp = ((size_t)z * H + y) * (size_t)W + x;
    stg_f(mask + sp, 1.f);
#pragma unroll
    for (int c = 0; c < 5; ++c)
        stg_u16(dense + sp * Cs + c, f2bf(vf[5 * i + c]));
}

// w (taps,Ci,Co) f32 -> wT (taps,Co,CiPad) bf16, zero-padded in ci
__global__ void prep_weights_k(const float* w, unsigned short* wt,
                               int taps, int Ci, int Co, int CiPad) {
    size_t i   = (size_t)blockIdx.x * blockDim.x + threadIdx.x;
    size_t tot = (size_t)taps * Co * CiPad;
    if (i >= tot) return;
    int    ci  = (int)(i % CiPad);
    size_t t   = i / CiPad;
    int    co  = (int)(t % Co);
    int    tap = (int)(t / Co);
    float  v   = (ci < Ci) ? w[((size_t)tap * Ci + ci) * Co + co] : 0.f;
    stg_u16(wt + i, f2bf(v));
}

__global__ void prep_bn_k(const float* gamma, const float* beta,
                          const float* mean, const float* var,
                          const float* bias, float* scale, float* shift, int C) {
    int c = blockIdx.x * blockDim.x + threadIdx.x;
    if (c >= C) return;
    float inv = gamma[c] / sqrtf(var[c] + 1e-3f);
    scale[c]  = inv;
    shift[c]  = beta[c] + (((bias != nullptr) ? bias[c] : 0.f) - mean[c]) * inv;
}

__global__ void down_mask_k(const float* mi, float* mo,
                            int Di, int Hi, int Wi, int Do, int Ho, int Wo,
                            int kd, int kh, int kw, int sd, int sh, int sw,
                            int pd, int ph, int pw) {
    size_t i   = (size_t)blockIdx.x * blockDim.x + threadIdx.x;
    size_t tot = (size_t)Do * Ho * Wo;
    if (i >= tot) return;
    int    x = (int)(i % Wo);
    size_t t = i / Wo;
    int    y = (int)(t % Ho);
    int    z = (int)(t / Ho);
    float  v = 0.f;
    for (int dz = 0; dz < kd; ++dz) {
        int zi = z * sd + dz - pd;
        if ((unsigned)zi >= (unsigned)Di) continue;
        for (int dy = 0; dy < kh; ++dy) {
            int yi = y * sh + dy - ph;
            if ((unsigned)yi >= (unsigned)Hi) continue;
            for (int dx = 0; dx < kw; ++dx) {
                int xi = x * sw + dx - pw;
                if ((unsigned)xi >= (unsigned)Wi) continue;
                if (ldg_f(mi + ((size_t)zi * Hi + yi) * (size_t)Wi + xi) > 0.f)
                    v = 1.f;
            }
        }
    }
    stg_f(mo + i, v);
}

// ---- host side -------------------------------------------------------------

static void run_conv(hipStream_t s, const ConvP& p) {
    int tilesX = (p.Wout + 15) / 16;
    int tiles  = tilesX * p.Hout * p.Dout;
    int gx     = (tiles + 3) / 4;
    dim3 blk(32, 4, 1);
    if (p.Co <= 16) {
        conv3d_wmma<1><<<dim3(gx, 1, 1), blk, 0, s>>>(p);
    } else if (p.Co == 32) {
        conv3d_wmma<2><<<dim3(gx, 1, 1), blk, 0, s>>>(p);
    } else {
        conv3d_wmma<4><<<dim3(gx, p.Co / 64, 1), blk, 0, s>>>(p);
    }
}

extern "C" void kernel_launch(void* const* d_in, const int* in_sizes, int n_in,
                              void* d_out, int out_size, void* d_ws, size_t ws_size,
                              hipStream_t stream) {
    (void)in_sizes; (void)n_in; (void)out_size; (void)ws_size;

    // Input leaf order: jax pytree flatten (dict keys sorted alphabetically):
    // 0:coors | params: b1a..b4b (each: b1,b2,bn1{beta,gamma,mean,var},
    // bn2{...},w1,w2 = 12 leaves), bn_d2,bn_d3,bn_d4,bn_ex,bn_in (4 each),
    // w_d2,w_d3,w_d4,w_ex,w_in | 122:voxel_features
    auto F = [&](int i) { return (const float*)d_in[i]; };
    const int*   coors = (const int*)d_in[0];
    const float* vfeat = (const float*)d_in[122];

    // workspace allocator (256B aligned)
    uintptr_t wbase = (uintptr_t)d_ws;
    size_t    cur   = 0;
    auto alloc = [&](size_t bytes) -> void* {
        void* p = (void*)(wbase + cur);
        cur += (bytes + 255) & ~(size_t)255;
        return p;
    };

    const int D1 = 41, H1 = 200, W1 = 200;
    const int D2 = 21, H2 = 100, W2 = 100;
    const int D3 = 11, H3 = 50,  W3 = 50;
    const int D4 = 5,  H4 = 25,  W4 = 25;
    const int D5 = 2,  H5 = 25,  W5 = 25;
    const size_t S1 = (size_t)D1 * H1 * W1;
    const size_t S2 = (size_t)D2 * H2 * W2;
    const size_t S3 = (size_t)D3 * H3 * W3;
    const size_t S4 = (size_t)D4 * H4 * W4;
    const size_t S5 = (size_t)D5 * H5 * W5;

    unsigned short* denseIn = (unsigned short*)alloc(S1 * 8 * 2);   // Cs=8 (5 + pad)
    unsigned short* bufA    = (unsigned short*)alloc(S1 * 16 * 2);
    unsigned short* bufB    = (unsigned short*)alloc(S1 * 16 * 2);
    unsigned short* bufC    = (unsigned short*)alloc(S1 * 16 * 2);
    float* mask1 = (float*)alloc(S1 * 4);
    float* mask2 = (float*)alloc(S2 * 4);
    float* mask3 = (float*)alloc(S3 * 4);
    float* mask4 = (float*)alloc(S4 * 4);
    float* mask5 = (float*)alloc(S5 * 4);

    // conv table: leaf indices (see mapping above)
    const int NC = 21;
    int wL[NC], ciA[NC], coA[NC], tapsA[NC], biasL[NC], bnL[NC], cipA[NC];
    int nC = 0;
    auto addc = [&](int w, int ci, int co, int taps, int bias, int bn) {
        wL[nC] = w; ciA[nC] = ci; coA[nC] = co; tapsA[nC] = taps;
        biasL[nC] = bias; bnL[nC] = bn;
        cipA[nC] = (ci <= 32) ? 32 : ci;
        ++nC;
    };
    addc(121,   5,  16, 27, -1, 113);            // 0  conv_in (bn_in)
    addc(11,   16,  16, 27,  1,   3);            // 1  b1a.w1
    addc(12,   16,  16, 27,  2,   7);            // 2  b1a.w2
    addc(23,   16,  16, 27, 13,  15);            // 3  b1b.w1
    addc(24,   16,  16, 27, 14,  19);            // 4  b1b.w2
    addc(117,  16,  32, 27, -1,  97);            // 5  w_d2 (bn_d2)
    addc(35,   32,  32, 27, 25,  27);            // 6  b2a.w1
    addc(36,   32,  32, 27, 26,  31);            // 7  b2a.w2
    addc(47,   32,  32, 27, 37,  39);            // 8  b2b.w1
    addc(48,   32,  32, 27, 38,  43);            // 9  b2b.w2
    addc(118,  32,  64, 27, -1, 101);            // 10 w_d3 (bn_d3)
    addc(59,   64,  64, 27, 49,  51);            // 11 b3a.w1
    addc(60,   64,  64, 27, 50,  55);            // 12 b3a.w2
    addc(71,   64,  64, 27, 61,  63);            // 13 b3b.w1
    addc(72,   64,  64, 27, 62,  67);            // 14 b3b.w2
    addc(119,  64, 128, 27, -1, 105);            // 15 w_d4 (bn_d4)
    addc(83,  128, 128, 27, 73,  75);            // 16 b4a.w1
    addc(84,  128, 128, 27, 74,  79);            // 17 b4a.w2
    addc(95,  128, 128, 27, 85,  87);            // 18 b4b.w1
    addc(96,  128, 128, 27, 86,  91);            // 19 b4b.w2
    addc(120, 128, 128,  3, -1, 109);            // 20 w_ex (bn_ex)

    unsigned short* wT[NC];
    float* scl[NC];
    float* shf[NC];
    for (int i = 0; i < NC; ++i) {
        size_t n = (size_t)tapsA[i] * coA[i] * cipA[i];
        wT[i]  = (unsigned short*)alloc(n * 2);
        scl[i] = (float*)alloc(coA[i] * 4);
        shf[i] = (float*)alloc(coA[i] * 4);
        prep_weights_k<<<dim3((unsigned)((n + 255) / 256)), dim3(256), 0, stream>>>(
            F(wL[i]), wT[i], tapsA[i], ciA[i], coA[i], cipA[i]);
        // bn leaves in order beta, gamma, mean, var
        prep_bn_k<<<dim3((coA[i] + 63) / 64), dim3(64), 0, stream>>>(
            F(bnL[i] + 1), F(bnL[i]), F(bnL[i] + 2), F(bnL[i] + 3),
            biasL[i] >= 0 ? F(biasL[i]) : nullptr, scl[i], shf[i], coA[i]);
    }

    // zero dense input + mask1 every call (graph-replay safe), then scatter
    {
        size_t nw = S1 * 4;   // S1*8 bf16 = S1*4 dwords
        zero_k<<<dim3((unsigned)((nw + 255) / 256)), dim3(256), 0, stream>>>(
            (unsigned int*)denseIn, nw);
        zero_k<<<dim3((unsigned)((S1 + 255) / 256)), dim3(256), 0, stream>>>(
            (unsigned int*)mask1, S1);
        scatter_k<<<dim3((60000 + 255) / 256), dim3(256), 0, stream>>>(
            vfeat, coors, denseIn, mask1, H1, W1, 8, 60000);
    }

    auto conv = [&](const unsigned short* in, int Di, int Hi, int Wi, int Csi,
                    int idx, int kd, int kh, int kw, int sd, int sh, int sw,
                    int pd, int ph, int pw,
                    unsigned short* out, float* outf,
                    int Do, int Ho, int Wo, int Cso,
                    const float* mask, const unsigned short* resid) {
        ConvP p;
        p.in = in; p.wt = wT[idx]; p.out = out; p.out_f32 = outf;
        p.mask = mask; p.resid = resid; p.scale = scl[idx]; p.shift = shf[idx];
        p.Din = Di; p.Hin = Hi; p.Win = Wi; p.Cs_in = Csi;
        p.Dout = Do; p.Hout = Ho; p.Wout = Wo; p.Cs_out = Cso;
        p.kd = kd; p.kh = kh; p.kw = kw; p.sd = sd; p.sh = sh; p.sw = sw;
        p.pd = pd; p.ph = ph; p.pw = pw;
        p.Co = coA[idx]; p.CiPad = cipA[idx]; p.relu = 1;
        run_conv(stream, p);
    };
    auto subm = [&](const unsigned short* in, int idx, unsigned short* out,
                    int D, int H, int W, int C, const float* mask,
                    const unsigned short* resid) {
        conv(in, D, H, W, C, idx, 3, 3, 3, 1, 1, 1, 1, 1, 1,
             out, nullptr, D, H, W, C, mask, resid);
    };
    auto dmask = [&](const float* mi, float* mo, int Di, int Hi, int Wi,
                     int Do, int Ho, int Wo, int kd, int kh, int kw,
                     int sd, int sh, int sw, int pd, int ph, int pw) {
        size_t tot = (size_t)Do * Ho * Wo;
        down_mask_k<<<dim3((unsigned)((tot + 255) / 256)), dim3(256), 0, stream>>>(
            mi, mo, Di, Hi, Wi, Do, Ho, Wo, kd, kh, kw, sd, sh, sw, pd, ph, pw);
    };

    // ---- stage 1 (41,200,200,16) ----
    conv(denseIn, D1, H1, W1, 8, 0, 3, 3, 3, 1, 1, 1, 1, 1, 1,
         bufA, nullptr, D1, H1, W1, 16, mask1, nullptr);          // conv_in
    subm(bufA, 1, bufB, D1, H1, W1, 16, mask1, nullptr);          // b1a.c1
    subm(bufB, 2, bufC, D1, H1, W1, 16, mask1, bufA);             // b1a.c2 (+res)
    subm(bufC, 3, bufA, D1, H1, W1, 16, mask1, nullptr);          // b1b.c1
    subm(bufA, 4, bufB, D1, H1, W1, 16, mask1, bufC);             // b1b.c2 (+res)

    // ---- down2 -> (21,100,100,32) ----
    dmask(mask1, mask2, D1, H1, W1, D2, H2, W2, 3, 3, 3, 2, 2, 2, 1, 1, 1);
    conv(bufB, D1, H1, W1, 16, 5, 3, 3, 3, 2, 2, 2, 1, 1, 1,
         bufA, nullptr, D2, H2, W2, 32, mask2, nullptr);
    subm(bufA, 6, bufB, D2, H2, W2, 32, mask2, nullptr);
    subm(bufB, 7, bufC, D2, H2, W2, 32, mask2, bufA);
    subm(bufC, 8, bufA, D2, H2, W2, 32, mask2, nullptr);
    subm(bufA, 9, bufB, D2, H2, W2, 32, mask2, bufC);

    // ---- down3 -> (11,50,50,64) ----
    dmask(mask2, mask3, D2, H2, W2, D3, H3, W3, 3, 3, 3, 2, 2, 2, 1, 1, 1);
    conv(bufB, D2, H2, W2, 32, 10, 3, 3, 3, 2, 2, 2, 1, 1, 1,
         bufA, nullptr, D3, H3, W3, 64, mask3, nullptr);
    subm(bufA, 11, bufB, D3, H3, W3, 64, mask3, nullptr);
    subm(bufB, 12, bufC, D3, H3, W3, 64, mask3, bufA);
    subm(bufC, 13, bufA, D3, H3, W3, 64, mask3, nullptr);
    subm(bufA, 14, bufB, D3, H3, W3, 64, mask3, bufC);

    // ---- down4 (pads 0,1,1) -> (5,25,25,128) ----
    dmask(mask3, mask4, D3, H3, W3, D4, H4, W4, 3, 3, 3, 2, 2, 2, 0, 1, 1);
    conv(bufB, D3, H3, W3, 64, 15, 3, 3, 3, 2, 2, 2, 0, 1, 1,
         bufA, nullptr, D4, H4, W4, 128, mask4, nullptr);
    subm(bufA, 16, bufB, D4, H4, W4, 128, mask4, nullptr);
    subm(bufB, 17, bufC, D4, H4, W4, 128, mask4, bufA);
    subm(bufC, 18, bufA, D4, H4, W4, 128, mask4, nullptr);
    subm(bufA, 19, bufB, D4, H4, W4, 128, mask4, bufC);

    // ---- ex conv (3,1,1) stride (2,1,1) -> f32 NCDHW (== (1,256,25,25)) ----
    dmask(mask4, mask5, D4, H4, W4, D5, H5, W5, 3, 1, 1, 2, 1, 1, 0, 0, 0);
    conv(bufB, D4, H4, W4, 128, 20, 3, 1, 1, 2, 1, 1, 0, 0, 0,
         nullptr, (float*)d_out, D5, H5, W5, 128, mask5, nullptr);
}
